// SelfAttention_49151605735684
// MI455X (gfx1250) — matrix operand order repaired and checked by reading, not compiled, not run on hardware
//
#include <hip/hip_runtime.h>
#include <hip/hip_bf16.h>
#include <math.h>

// ---------------- CDNA5 WMMA types ----------------
typedef __attribute__((ext_vector_type(16))) __bf16 v16bf;
typedef __attribute__((ext_vector_type(8)))  float  v8f;

union ABu { v16bf v; __bf16 e[16]; uint4 q[2]; };
union V8u { v8f v; float f[8]; };

#define WMMA_BF16(a, b, c) \
  __builtin_amdgcn_wmma_f32_16x16x32_bf16(false, (a), false, (b), (short)0, (c), false, false)

// Problem constants (from reference setup_inputs)
constexpr int Bb = 8;      // batch
constexpr int Cc = 256;    // channels
constexpr int Nn = 4096;   // W*H
constexpr int CQ = 32;     // C/8  (q,k head dim)
constexpr int CV = 128;    // C/2  (v dim)

// ---------------- Phase 0a: transpose + f32->bf16:  x[B,C,N] -> xt[B,N,C] ----------------
__global__ void __launch_bounds__(256) xpose_bf16(const float* __restrict__ x,
                                                  __bf16* __restrict__ xt) {
  __shared__ float tile[32][33];
  const int b  = blockIdx.z;
  const int n0 = blockIdx.x * 32;
  const int c0 = blockIdx.y * 32;
  const int tx = threadIdx.x;   // 0..31
  const int ty = threadIdx.y;   // 0..7
  const float* xb = x + (size_t)b * Cc * Nn;
  for (int k = 0; k < 32; k += 8)
    tile[ty + k][tx] = xb[(size_t)(c0 + ty + k) * Nn + n0 + tx];
  __syncthreads();
  __bf16* xtb = xt + (size_t)b * Nn * Cc;
  for (int k = 0; k < 32; k += 8)
    xtb[(size_t)(n0 + ty + k) * Cc + c0 + tx] = (__bf16)tile[tx][ty + k];
}

// ---------------- Phase 0b: f32 -> bf16 convert (weights) ----------------
__global__ void __launch_bounds__(256) cvt_bf16(const float* __restrict__ in,
                                                __bf16* __restrict__ out, int n) {
  int i = blockIdx.x * 256 + threadIdx.x;
  if (i < n) out[i] = (__bf16)in[i];
}

// ---------------- Phase 1: QKV projections, one wave per 16x16 tile ----------------
// q[b,n,o] = sum_c xt[b,n,c] * W[o,c] + bias[o]
__global__ void __launch_bounds__(32) proj_qkv(
    const __bf16* __restrict__ xt,
    const __bf16* __restrict__ wqb, const __bf16* __restrict__ wkb,
    const __bf16* __restrict__ wvb,
    const float* __restrict__ bq, const float* __restrict__ bk,
    const float* __restrict__ bv,
    __bf16* __restrict__ qb, __bf16* __restrict__ kb, __bf16* __restrict__ vb)
{
  const int t  = blockIdx.x % 12;               // 16-wide output tile (2 q, 2 k, 8 v)
  const int nt = (blockIdx.x / 12) & 255;       // N/16 tiles
  const int b  = blockIdx.x / (12 * 256);
  const int n0 = nt << 4;
  const int lane = threadIdx.x;
  const int col = lane & 15, hi = lane >> 4, koff = hi * 8;

  const __bf16* W; const float* bias; __bf16* dst; int obase, mode;
  if (t < 2)      { W = wqb; bias = bq; obase = t * 16;       dst = qb; mode = 0; }
  else if (t < 4) { W = wkb; bias = bk; obase = (t - 2) * 16; dst = kb; mode = 0; }
  else            { W = wvb; bias = bv; obase = (t - 4) * 16; dst = vb; mode = 1; }

  V8u acc;
  const float b0 = bias[obase + col];
  for (int e = 0; e < 8; ++e) acc.f[e] = b0;

  const __bf16* arow = xt + ((size_t)b * Nn + n0 + col) * Cc;     // A: row n, contiguous c
  const __bf16* brow = W + (size_t)(obase + col) * Cc + hi * 8;   // B: col o, K=c in table order
  for (int cc = 0; cc < Cc / 32; ++cc) {
    ABu a, bm;
    a.q[0]  = *(const uint4*)(arow + cc * 32 + koff);
    a.q[1]  = *(const uint4*)(arow + cc * 32 + 16 + koff);
    bm.q[0] = *(const uint4*)(brow + cc * 32);
    bm.q[1] = *(const uint4*)(brow + cc * 32 + 16);
    acc.v = WMMA_BF16(a.v, bm.v, acc.v);
  }
  if (mode == 0) {                       // q,k: [B,N,32]
    for (int e = 0; e < 8; ++e)
      dst[((size_t)b * Nn + n0 + e + 8 * hi) * CQ + obase + col] = (__bf16)acc.f[e];
  } else {                               // v: [B,128,N]
    for (int e = 0; e < 8; ++e)
      dst[((size_t)b * CV + obase + col) * Nn + n0 + e + 8 * hi] = (__bf16)acc.f[e];
  }
}

// ---------------- Phase 2: flash attention ----------------
// 8 waves / block share one K/V stream via LDS; each wave owns a 16-query tile.
__global__ void __launch_bounds__(256) flash_attn(
    const __bf16* __restrict__ qb, const __bf16* __restrict__ kb,
    const __bf16* __restrict__ vb, __bf16* __restrict__ ot)
{
  __shared__ __align__(16) __bf16 Kl[32 * 32];       // [j_local][d]  2 KB
  __shared__ __align__(16) __bf16 Vl[128 * 32];      // [c][j_local]  8 KB
  __shared__ __align__(16) __bf16 Pl[8 * 16 * 32];   // per-wave P    8 KB

  const int b   = blockIdx.x >> 5;            // 32 blocks per batch
  const int qb0 = (blockIdx.x & 31) << 7;     // 128 queries per block
  const int tid = threadIdx.x;
  const int w    = tid >> 5;                  // wave id 0..7
  const int lane = tid & 31;
  const int col = lane & 15, hi = lane >> 4, koff = hi * 8;
  const int i0 = qb0 + w * 16;                // this wave's query tile

  // cooperative-load indexing (256 threads)
  const int kj = tid >> 3;                    // 0..31 K-tile row
  const int kd = (tid & 7) * 4;               // 4 bf16 per thread
  const int vc = tid >> 1;                    // 0..127 V channel
  const int vh = (tid & 1) * 16;              // j half

  __bf16* plw = &Pl[w * 16 * 32];

  // Q as WMMA A operand (16 rows x K=32)
  ABu qa;
  const __bf16* qrow = qb + ((size_t)b * Nn + i0 + col) * CQ;
  qa.q[0] = *(const uint4*)(qrow + koff);
  qa.q[1] = *(const uint4*)(qrow + 16 + koff);

  V8u oacc[8];
  float lsum[8], mrow[8];
  for (int t = 0; t < 8; ++t)
    for (int e = 0; e < 8; ++e) oacc[t].f[e] = 0.f;
  for (int e = 0; e < 8; ++e) { lsum[e] = 0.f; mrow[e] = -1e30f; }

  const __bf16* kbase = kb + (size_t)b * Nn * CQ;
  const __bf16* vbase = vb + (size_t)b * CV * Nn;

  for (int j0 = 0; j0 < Nn; j0 += 32) {
    __syncthreads();   // previous iteration's LDS reads complete before overwrite
    // stage K tile: Kl[jl][d]
    *(uint2*)&Kl[kj * 32 + kd] = *(const uint2*)(kbase + (size_t)(j0 + kj) * CQ + kd);
    // stage V tile: Vl[c][jl]
    {
      const __bf16* vsrc = vbase + (size_t)vc * Nn + j0 + vh;
      *(uint4*)&Vl[vc * 32 + vh]     = *(const uint4*)(vsrc);
      *(uint4*)&Vl[vc * 32 + vh + 8] = *(const uint4*)(vsrc + 8);
    }
    __syncthreads();

    // K^T as B operand from LDS: B[d][j] = Kl[j][d], d in table order per lane
    ABu kt0, kt1;
    const __bf16* kr0 = &Kl[col * 32 + hi * 8];
    const __bf16* kr1 = &Kl[(16 + col) * 32 + hi * 8];
    kt0.q[0] = *(const uint4*)(kr0);
    kt0.q[1] = *(const uint4*)(kr0 + 16);
    kt1.q[0] = *(const uint4*)(kr1);
    kt1.q[1] = *(const uint4*)(kr1 + 16);

    V8u s0, s1;
    for (int e = 0; e < 8; ++e) { s0.f[e] = 0.f; s1.f[e] = 0.f; }
    s0.v = WMMA_BF16(qa.v, kt0.v, s0.v);   // S[:, j0..j0+15]
    s1.v = WMMA_BF16(qa.v, kt1.v, s1.v);   // S[:, j0+16..j0+31]

    // online softmax: lane holds one key column across 8 rows (e + 8*hi)
    float p0[8], p1[8], alpha[8];
    for (int e = 0; e < 8; ++e) {
      float mx = fmaxf(s0.f[e], s1.f[e]);
      for (int off = 1; off < 16; off <<= 1)      // reduce across 16 key columns
        mx = fmaxf(mx, __shfl_xor(mx, off, 32));
      const float mn = fmaxf(mrow[e], mx);
      const float a  = __expf(mrow[e] - mn);
      p0[e] = __expf(s0.f[e] - mn);
      p1[e] = __expf(s1.f[e] - mn);
      float rs = p0[e] + p1[e];
      for (int off = 1; off < 16; off <<= 1)
        rs += __shfl_xor(rs, off, 32);
      lsum[e] = lsum[e] * a + rs;
      mrow[e] = mn;
      alpha[e] = a;
    }
    for (int t = 0; t < 8; ++t)
      for (int e = 0; e < 8; ++e) oacc[t].f[e] *= alpha[e];

    // P: C-layout -> A-layout via this wave's LDS region
    for (int e = 0; e < 8; ++e) {
      const int r = e + 8 * hi;
      plw[r * 32 + col]      = (__bf16)p0[e];
      plw[r * 32 + 16 + col] = (__bf16)p1[e];
    }
    __syncthreads();
    ABu pa;
    const __bf16* prow = &plw[col * 32];
    pa.q[0] = *(const uint4*)(prow + koff);
    pa.q[1] = *(const uint4*)(prow + 16 + koff);

    // O += P(16x32) @ V(32x128) from LDS
    for (int t = 0; t < 8; ++t) {
      ABu vt;
      const __bf16* vrow = &Vl[(t * 16 + col) * 32 + hi * 8];
      vt.q[0] = *(const uint4*)(vrow);
      vt.q[1] = *(const uint4*)(vrow + 16);
      oacc[t].v = WMMA_BF16(pa.v, vt.v, oacc[t].v);
    }
  }

  // normalize and store O as [B,N,128]
  for (int e = 0; e < 8; ++e) {
    const float inv = 1.f / lsum[e];
    const size_t row = (size_t)b * Nn + i0 + e + 8 * hi;
    for (int t = 0; t < 8; ++t)
      ot[row * CV + t * 16 + col] = (__bf16)(oacc[t].f[e] * inv);
  }
}

// ---------------- Phase 3: output projection + gamma*out + x ----------------
__global__ void __launch_bounds__(32) out_proj(
    const __bf16* __restrict__ ot, const __bf16* __restrict__ wob,
    const float* __restrict__ bo, const float* __restrict__ x,
    const float* __restrict__ gamma, float* __restrict__ y)
{
  const int ct = blockIdx.x & 15;            // C/16 output tiles
  const int nt = (blockIdx.x >> 4) & 255;
  const int b  = blockIdx.x >> 12;
  const int n0 = nt << 4;
  const int cobase = ct << 4;
  const int lane = threadIdx.x;
  const int col = lane & 15, hi = lane >> 4, koff = hi * 8;

  V8u acc;
  const float b0 = bo[cobase + col];
  for (int e = 0; e < 8; ++e) acc.f[e] = b0;

  const __bf16* arow = ot + ((size_t)b * Nn + n0 + col) * CV;
  const __bf16* brow = wob + (size_t)(cobase + col) * CV + hi * 8;
  for (int cc = 0; cc < CV / 32; ++cc) {
    ABu a, bm;
    a.q[0]  = *(const uint4*)(arow + cc * 32 + koff);
    a.q[1]  = *(const uint4*)(arow + cc * 32 + 16 + koff);
    bm.q[0] = *(const uint4*)(brow + cc * 32);
    bm.q[1] = *(const uint4*)(brow + cc * 32 + 16);
    acc.v = WMMA_BF16(a.v, bm.v, acc.v);
  }
  const float g = gamma[0];
  for (int e = 0; e < 8; ++e) {
    const size_t idx = ((size_t)b * Cc + cobase + col) * Nn + n0 + e + 8 * hi;
    y[idx] = g * acc.f[e] + x[idx];
  }
}

// ---------------- Host launcher ----------------
extern "C" void kernel_launch(void* const* d_in, const int* in_sizes, int n_in,
                              void* d_out, int out_size, void* d_ws, size_t ws_size,
                              hipStream_t stream) {
  (void)in_sizes; (void)n_in; (void)out_size; (void)ws_size;
  const float* x     = (const float*)d_in[0];
  const float* Wq    = (const float*)d_in[1];
  const float* bq    = (const float*)d_in[2];
  const float* Wk    = (const float*)d_in[3];
  const float* bk    = (const float*)d_in[4];
  const float* Wv    = (const float*)d_in[5];
  const float* bv    = (const float*)d_in[6];
  const float* Wo    = (const float*)d_in[7];
  const float* bo    = (const float*)d_in[8];
  const float* gamma = (const float*)d_in[9];
  float* y = (float*)d_out;

  char* ws = (char*)d_ws;
  size_t off = 0;
  __bf16* xt  = (__bf16*)(ws + off); off += (size_t)Bb * Nn * Cc * 2;  // 16.78 MB
  __bf16* qbf = (__bf16*)(ws + off); off += (size_t)Bb * Nn * CQ * 2;  //  2.10 MB
  __bf16* kbf = (__bf16*)(ws + off); off += (size_t)Bb * Nn * CQ * 2;  //  2.10 MB
  __bf16* vbf = (__bf16*)(ws + off); off += (size_t)Bb * CV * Nn * 2;  //  8.39 MB
  __bf16* otb = (__bf16*)(ws + off); off += (size_t)Bb * Nn * CV * 2;  //  8.39 MB
  __bf16* wqb = (__bf16*)(ws + off); off += (size_t)CQ * Cc * 2;
  __bf16* wkb = (__bf16*)(ws + off); off += (size_t)CQ * Cc * 2;
  __bf16* wvb = (__bf16*)(ws + off); off += (size_t)CV * Cc * 2;
  __bf16* wob = (__bf16*)(ws + off); off += (size_t)Cc * CV * 2;

  // Phase 0: transpose x + convert weights to bf16
  xpose_bf16<<<dim3(Nn / 32, Cc / 32, Bb), dim3(32, 8), 0, stream>>>(x, xt);
  cvt_bf16<<<(CQ * Cc + 255) / 256, 256, 0, stream>>>(Wq, wqb, CQ * Cc);
  cvt_bf16<<<(CQ * Cc + 255) / 256, 256, 0, stream>>>(Wk, wkb, CQ * Cc);
  cvt_bf16<<<(CV * Cc + 255) / 256, 256, 0, stream>>>(Wv, wvb, CV * Cc);
  cvt_bf16<<<(Cc * CV + 255) / 256, 256, 0, stream>>>(Wo, wob, Cc * CV);

  // Phase 1: QKV projections (WMMA)
  proj_qkv<<<Bb * 256 * 12, 32, 0, stream>>>(xt, wqb, wkb, wvb, bq, bk, bv,
                                             qbf, kbf, vbf);
  // Phase 2: flash attention (WMMA, LDS-shared K/V)
  flash_attn<<<Bb * 32, 256, 0, stream>>>(qbf, kbf, vbf, otb);

  // Phase 3: output projection + residual (WMMA)
  out_proj<<<Bb * 256 * 16, 32, 0, stream>>>(otb, wob, bo, x, gamma, y);
}
